// LINESG_88699664597071
// MI455X (gfx1250) — matrix operand order repaired and checked by reading, not compiled, not run on hardware
//
#include <hip/hip_runtime.h>
#include <hip/hip_bf16.h>

// ---------------------------------------------------------------------------
// Problem constants (from reference)
// ---------------------------------------------------------------------------
#define N_USERS   16000
#define LATENT    64
#define E_ORIG    200000
#define K_CAND    50000
#define N_PAIRS   (E_ORIG + K_CAND)      // 250000
#define COS_THR   0.5f
#define TAU       0.2f

// candidate capture threshold (below COS_THR so top-K can dip under 0.5)
#define CAND_THR  0.40f
#define CAND_CAP  (1u << 22)             // 4M candidate slots
#define BINS      2048

typedef __attribute__((ext_vector_type(16))) __bf16        v16bf;
typedef __attribute__((ext_vector_type(8)))  float         v8f;
typedef __attribute__((ext_vector_type(8)))  unsigned int  v8u;
typedef __attribute__((ext_vector_type(4)))  unsigned int  u32x4;
typedef __attribute__((ext_vector_type(4)))  int           i32x4;
typedef __attribute__((ext_vector_type(8)))  int           i32x8;

// ---------------------------------------------------------------------------
// K1: row-normalize user_emb -> bf16   (16000 rows, 64 threads per row)
// ---------------------------------------------------------------------------
__global__ void k_norm(const float* __restrict__ emb, __hip_bfloat16* __restrict__ u)
{
    int row = blockIdx.x;
    int c   = threadIdx.x;                  // 0..63
    float x = emb[(size_t)row * LATENT + c];
    __shared__ float red[64];
    red[c] = x * x;
    __syncthreads();
    for (int s = 32; s > 0; s >>= 1) {
        if (c < s) red[c] += red[c + s];
        __syncthreads();
    }
    float nrm = fmaxf(sqrtf(red[0]), 1e-12f);
    u[(size_t)row * LATENT + c] = __float2bfloat16(x / nrm);
}

// ---------------------------------------------------------------------------
// K2: mark existing edges in a 16000x16000 bitmap (directed, as in reference)
// ---------------------------------------------------------------------------
__global__ void k_edges(const int* __restrict__ nz, unsigned int* __restrict__ bitmap)
{
    int e = blockIdx.x * blockDim.x + threadIdx.x;
    if (e >= E_ORIG) return;
    int i = nz[2 * e], j = nz[2 * e + 1];
    size_t bit = (size_t)i * N_USERS + (size_t)j;
    atomicOr(&bitmap[bit >> 5], 1u << (bit & 31));
}

// ---------------------------------------------------------------------------
// K3: WMMA cos-similarity GEMM (u @ u^T, bf16 in / f32 acc) with fused
//     threshold filter.  Block = 8 waves -> 128x128 C tile.  The 128 B-side
//     rows (16 KB, contiguous) are staged into LDS by the Tensor Data Mover
//     (one tensor_load_to_lds issued by wave 0, TENSORcnt-tracked), then each
//     wave runs 16 v_wmma_f32_16x16x32_bf16 with A-fragment reuse.
// ---------------------------------------------------------------------------
__global__ __launch_bounds__(256)
void k_gemm(const unsigned int* __restrict__ U,        // u_bf16 as dwords, 32/row
            const unsigned int* __restrict__ bitmap,
            float*        __restrict__ cand_val,
            unsigned int* __restrict__ cand_idx,
            int*          __restrict__ cnt)
{
    __shared__ unsigned int Bs[128 * 32];              // 128 rows x 64 bf16

    const int tm = blockIdx.x * 128;
    const int tn = blockIdx.y * 128;

    // ---- TDM stage of the 128 B-side rows (C columns) into LDS ------------
    if (threadIdx.x < 32) {                            // one wave issues DMA
        unsigned long long ga =
            (unsigned long long)(const char*)(U + (size_t)tn * 32);
        u32x4 g0;
        g0[0] = 1u;                                    // count=1 (valid D#)
        g0[1] = (unsigned int)(size_t)&Bs[0];          // lds_addr (bytes)
        g0[2] = (unsigned int)(ga & 0xFFFFFFFFu);      // global_addr[31:0]
        g0[3] = (unsigned int)((ga >> 32) & 0x01FFFFFFu) | (2u << 30); // type=2
        i32x8 g1 = {};
        g1[0] = 0x00020000;                            // data_size = 4 bytes
        g1[1] = (int)(32u << 16);                      // tensor_dim0 = 32 dw
        g1[2] = (int)((unsigned)N_USERS << 16);        // tensor_dim1 = 16000
        g1[3] = (int)(32u << 16);                      // tile_dim0 = 32 dw
        g1[4] = 128;                                   // tile_dim1 = 128 rows
        g1[5] = 32;                                    // dim0 stride = 32 dw
        g1[6] = 0;
        g1[7] = 0;
        i32x4 z4 = {};
#if __clang_major__ >= 23
        i32x8 z8 = {};
        __builtin_amdgcn_tensor_load_to_lds(g0, g1, z4, z4, z8, 0);
#else
        __builtin_amdgcn_tensor_load_to_lds(g0, g1, z4, z4, 0);
#endif
        __builtin_amdgcn_s_wait_tensorcnt(0);
    }
    __syncthreads();

    const int wave  = threadIdx.x >> 5;                // 0..7 -> 16-row strip
    const int lane  = threadIdx.x & 31;
    const int lrow  = lane & 15;
    const int lhalf = lane >> 4;

    // ---- A fragments (rows tm + wave*16 + lrow), K 0..31 and 32..63 -------
    // 16-bit A 16x32 layout: VGPR v, half h -> k = (v<4?0:16) + (v&3)*2 + h*8
    const unsigned int* ap = U + (size_t)(tm + wave * 16 + lrow) * 32;
    v8u au0, au1;
#pragma unroll
    for (int v = 0; v < 8; ++v) {
        int d = ((v & 4) << 1) + (v & 3) + (lhalf << 2);   // dword idx, frag0
        au0[v] = ap[d];
        au1[v] = ap[d + 16];                               // +32 bf16
    }
    const v16bf a0 = __builtin_bit_cast(v16bf, au0);
    const v16bf a1 = __builtin_bit_cast(v16bf, au1);

#pragma unroll
    for (int nt = 0; nt < 8; ++nt) {
        // ---- B fragment for C columns tn + nt*16 .. +15 from LDS ----------
        // 16-bit B 32x16 layout: same K order as A (VGPR v, half h -> k = (v<4?0:16) + (v&3)*2 + h*8)
        const int brow = nt * 16 + lrow;                   // LDS row = C col
        v8u bu0, bu1;
#pragma unroll
        for (int v = 0; v < 8; ++v) {
            int d = ((v & 4) << 1) + (v & 3) + (lhalf << 2); // dword idx, frag0
            bu0[v] = Bs[brow * 32 + d];
            bu1[v] = Bs[brow * 32 + d + 16];
        }
        v16bf b0 = __builtin_bit_cast(v16bf, bu0);
        v16bf b1 = __builtin_bit_cast(v16bf, bu1);

        v8f c = {};
        c = __builtin_amdgcn_wmma_f32_16x16x32_bf16(false, a0, false, b0,
                                                    (short)0, c, false, false);
        c = __builtin_amdgcn_wmma_f32_16x16x32_bf16(false, a1, false, b1,
                                                    (short)0, c, false, false);

        // ---- fused epilogue: wave-level gate, then rare slow path ---------
        float mx = fmaxf(fmaxf(fmaxf(c[0], c[1]), fmaxf(c[2], c[3])),
                         fmaxf(fmaxf(c[4], c[5]), fmaxf(c[6], c[7])));
        if (__any(mx > CAND_THR)) {
            const int ccol = tn + nt * 16 + lrow;          // C: N = lane&15
#pragma unroll
            for (int j = 0; j < 8; ++j) {                  // C: M = j+8*half
                float val = c[j];
                if (val > CAND_THR) {
                    int crow = tm + wave * 16 + j + 8 * lhalf;
                    if (crow != ccol) {
                        size_t bit = (size_t)crow * N_USERS + (size_t)ccol;
                        if (!((bitmap[bit >> 5] >> (bit & 31)) & 1u)) {
                            unsigned int pos =
                                (unsigned int)atomicAdd(&cnt[0], 1);
                            if (pos < CAND_CAP) {
                                cand_val[pos] = val;
                                cand_idx[pos] = (unsigned int)bit;
                            }
                        }
                    }
                }
            }
        }
    }
}

// ---------------------------------------------------------------------------
// K4..K7: histogram-based top-K selection over the filtered candidates
// ---------------------------------------------------------------------------
__device__ __forceinline__ int bin_of(float v)
{
    int b = (int)((v - CAND_THR) * (BINS / (1.0f - CAND_THR)));
    return b < 0 ? 0 : (b > BINS - 1 ? BINS - 1 : b);
}

__global__ void k_hist(const float* __restrict__ cand_val, const int* __restrict__ cnt,
                       int* __restrict__ hist)
{
    int i = blockIdx.x * blockDim.x + threadIdx.x;
    int n = cnt[0]; if (n > (int)CAND_CAP) n = CAND_CAP;
    if (i >= n) return;
    atomicAdd(&hist[bin_of(cand_val[i])], 1);
}

__global__ void k_cut(const int* __restrict__ hist, int* __restrict__ cnt)
{
    long long acc = 0;
    int cut = -1, take = 0;
    for (int b = BINS - 1; b >= 0; --b) {
        long long h = hist[b];
        if (acc + h >= (long long)K_CAND) { cut = b; take = (int)(K_CAND - acc); break; }
        acc += h;
    }
    cnt[1] = cut;
    cnt[2] = take;
}

__global__ void k_compact(const float* __restrict__ cand_val,
                          const unsigned int* __restrict__ cand_idx,
                          int* __restrict__ cnt,
                          float* __restrict__ sel_val, unsigned int* __restrict__ sel_idx)
{
    int i = blockIdx.x * blockDim.x + threadIdx.x;
    int n = cnt[0]; if (n > (int)CAND_CAP) n = CAND_CAP;
    if (i >= n) return;
    float v  = cand_val[i];
    int   b  = bin_of(v);
    int   cut = cnt[1];
    int   pos = -1;
    if (b > cut) {
        pos = atomicAdd(&cnt[3], 1);
    } else if (b == cut) {
        int t = atomicAdd(&cnt[4], 1);
        if (t < cnt[2]) pos = atomicAdd(&cnt[3], 1);
    }
    if (pos >= 0 && pos < K_CAND) { sel_val[pos] = v; sel_idx[pos] = cand_idx[i]; }
}

__global__ void k_fill(const int* __restrict__ cnt,
                       float* __restrict__ sel_val, unsigned int* __restrict__ sel_idx)
{
    int s = blockIdx.x * blockDim.x + threadIdx.x;
    if (s >= K_CAND) return;
    if (s >= cnt[3]) { sel_val[s] = -1.0f; sel_idx[s] = 0u; }
}

// ---------------------------------------------------------------------------
// K8: build pair list (edges + candidates), per-pair base weight, and write
//     the pair_idx output region of d_out.
// ---------------------------------------------------------------------------
__global__ void k_pairs(const int* __restrict__ nz,
                        const float* __restrict__ sel_val,
                        const unsigned int* __restrict__ sel_idx,
                        int* __restrict__ pairs, float* __restrict__ baseArr,
                        char* __restrict__ pair_out, int mode /*1=int64, 0=int32*/)
{
    int p = blockIdx.x * blockDim.x + threadIdx.x;
    if (p >= N_PAIRS) return;
    int a, b2; float base;
    if (p < E_ORIG) {
        a = nz[2 * p]; b2 = nz[2 * p + 1]; base = 1.0f;
    } else {
        int s = p - E_ORIG;
        unsigned int idx = sel_idx[s];
        a  = (int)(idx / N_USERS);
        b2 = (int)(idx % N_USERS);
        float v = sel_val[s];
        base = (v > COS_THR) ? v : 0.0f;   // max(where(v>thr, v, -1), 0)
    }
    pairs[2 * p] = a; pairs[2 * p + 1] = b2;
    baseArr[p] = base;
    if (mode == 1) {
        long long* o = (long long*)pair_out;
        o[2 * p] = a; o[2 * p + 1] = b2;
    } else {
        int* o = (int*)pair_out;
        o[2 * p] = a; o[2 * p + 1] = b2;
    }
}

// ---------------------------------------------------------------------------
// K9: pair MLP + gumbel straight-through.  One wave32 per pair.
// ---------------------------------------------------------------------------
__global__ __launch_bounds__(256)
void k_mlp(const float* __restrict__ emb, const float* __restrict__ W,
           const float* __restrict__ bvec, const float* __restrict__ gn,
           const int* __restrict__ pairs, const float* __restrict__ baseArr,
           float* __restrict__ out)
{
    int wid  = (blockIdx.x * blockDim.x + threadIdx.x) >> 5;   // pair id
    int lane = threadIdx.x & 31;
    if (wid >= N_PAIRS) return;

    int ua = pairs[2 * wid], va = pairs[2 * wid + 1];
    float p0 = 0.f, p1 = 0.f;
#pragma unroll
    for (int t = 0; t < 4; ++t) {
        int i = lane + 32 * t;                                  // 0..127
        float x = (i < LATENT) ? emb[(size_t)ua * LATENT + i]
                               : emb[(size_t)va * LATENT + (i - LATENT)];
        p0 = fmaf(x, W[2 * i],     p0);
        p1 = fmaf(x, W[2 * i + 1], p1);
    }
#pragma unroll
    for (int off = 16; off > 0; off >>= 1) {
        p0 += __shfl_xor(p0, off, 32);
        p1 += __shfl_xor(p1, off, 32);
    }
    if (lane == 0) {
        float y0 = (p0 + bvec[0] + gn[2 * wid])     * (1.0f / TAU);
        float y1 = (p1 + bvec[1] + gn[2 * wid + 1]) * (1.0f / TAU);
        float m  = fmaxf(y0, y1);
        float e0 = __expf(y0 - m), e1 = __expf(y1 - m);
        float soft0 = e0 / (e0 + e1);
        float hard0 = (y0 >= y1) ? 1.0f : 0.0f;      // argmax==0 on tie
        float g     = (hard0 - soft0) + soft0;       // straight-through fwd
        float g01   = g * baseArr[wid];
        out[wid]                = g01;               // gumbel_01
        out[N_PAIRS + wid]      = g01;               // gumbel_retain_w[0]
        out[2 * N_PAIRS + wid]  = g01;               // gumbel_retain_w[1]
    }
}

// ---------------------------------------------------------------------------
// Host orchestration
// ---------------------------------------------------------------------------
extern "C" void kernel_launch(void* const* d_in, const int* in_sizes, int n_in,
                              void* d_out, int out_size, void* d_ws, size_t ws_size,
                              hipStream_t stream)
{
    const float* emb = (const float*)d_in[0];    // (16000, 64)
    const int*   nz  = (const int*)d_in[1];      // (200000, 2)
    const float* W   = (const float*)d_in[2];    // (128, 2)
    const float* bv  = (const float*)d_in[3];    // (2,)
    const float* gn  = (const float*)d_in[4];    // (250000, 2)

    // workspace layout (bytes)
    char*  ws      = (char*)d_ws;
    size_t o_u     = 0;                                   // bf16 u: 2,048,000
    size_t o_bm    = o_u  + 2048000;                      // bitmap: 32,000,000
    size_t o_cv    = o_bm + 32000000;                     // cand vals: 16 MB
    size_t o_ci    = o_cv + (size_t)CAND_CAP * 4;         // cand idx : 16 MB
    size_t o_hist  = o_ci + (size_t)CAND_CAP * 4;         // hist: 8 KB
    size_t o_cnt   = o_hist + BINS * 4;                   // counters: 256 B
    size_t o_sv    = o_cnt + 256;                         // sel vals: 200 KB
    size_t o_si    = o_sv + (size_t)K_CAND * 4;           // sel idx : 200 KB
    size_t o_pairs = o_si + (size_t)K_CAND * 4;           // pairs i32: 2 MB
    size_t o_base  = o_pairs + (size_t)N_PAIRS * 2 * 4;   // base: 1 MB

    __hip_bfloat16* u_bf  = (__hip_bfloat16*)(ws + o_u);
    unsigned int*   bm    = (unsigned int*)(ws + o_bm);
    float*          cv    = (float*)(ws + o_cv);
    unsigned int*   ci    = (unsigned int*)(ws + o_ci);
    int*            hist  = (int*)(ws + o_hist);
    int*            cnt   = (int*)(ws + o_cnt);
    float*          sv    = (float*)(ws + o_sv);
    unsigned int*   si    = (unsigned int*)(ws + o_si);
    int*            pairs = (int*)(ws + o_pairs);
    float*          baseA = (float*)(ws + o_base);

    // zero mutable scratch each call (deterministic, graph-capturable)
    hipMemsetAsync(bm,   0, 32000000, stream);
    hipMemsetAsync(hist, 0, BINS * 4, stream);
    hipMemsetAsync(cnt,  0, 256,      stream);

    // pair_idx output encoding: int64 occupies 2 float-slots each
    int mode = (out_size - 3 * N_PAIRS == 2 * N_PAIRS * 2) ? 1 : 0;
    char* pair_out = (char*)d_out + (size_t)3 * N_PAIRS * 4;

    k_norm <<<N_USERS, 64, 0, stream>>>(emb, u_bf);
    k_edges<<<(E_ORIG + 255) / 256, 256, 0, stream>>>(nz, bm);
    k_gemm <<<dim3(N_USERS / 128, N_USERS / 128), 256, 0, stream>>>(
        (const unsigned int*)u_bf, bm, cv, ci, cnt);
    k_hist   <<<CAND_CAP / 256, 256, 0, stream>>>(cv, cnt, hist);
    k_cut    <<<1, 1, 0, stream>>>(hist, cnt);
    k_compact<<<CAND_CAP / 256, 256, 0, stream>>>(cv, ci, cnt, sv, si);
    k_fill   <<<(K_CAND + 255) / 256, 256, 0, stream>>>(cnt, sv, si);
    k_pairs  <<<(N_PAIRS + 255) / 256, 256, 0, stream>>>(nz, sv, si, pairs, baseA,
                                                         pair_out, mode);
    k_mlp    <<<(N_PAIRS * 32 + 255) / 256, 256, 0, stream>>>(emb, W, bv, gn, pairs,
                                                              baseA, (float*)d_out);
}
